// KGEModel_47261820125521
// MI455X (gfx1250) — matrix operand order repaired and checked by reading, not compiled, not run on hardware
//
#include <hip/hip_runtime.h>
#include <hip/hip_bf16.h>

typedef __attribute__((ext_vector_type(16))) _Float16 v16h;
typedef __attribute__((ext_vector_type(8)))  float    v8f;

#define EDIM 1024
#define SEQ_LD 1032   // halves; 2064B rows -> rotates banks
#define QK_LD  136    // 272B rows
#define VT_LD  32
#define P_LD   32

static __device__ __forceinline__ v8f wmma_f16(v16h a, v16h b, v8f c) {
  return __builtin_amdgcn_wmma_f32_16x16x32_f16(false, a, false, b, (short)0, c, false, false);
}

// A fragment: 16x32 f16 tile, row-major [row][k] with row stride `ld` halves.
static __device__ __forceinline__ v16h ldA(const _Float16* base, int row0, int k0, int ld) {
  int lane = threadIdx.x & 31; int r = lane & 15; int hi = lane >> 4;
  const _Float16* p = base + (row0 + r) * ld + k0 + hi * 8;
  v16h a;
#pragma unroll
  for (int t = 0; t < 8; ++t) a[t] = p[t];
#pragma unroll
  for (int t = 0; t < 8; ++t) a[8 + t] = p[16 + t];
  return a;
}

// B fragment: 32x16 f16 tile of W^T stored [n][k] (contiguous K, stride ld halves).
static __device__ __forceinline__ v16h ldB(const _Float16* base, int n0, int k0, int ld) {
  int lane = threadIdx.x & 31; int n = lane & 15; int hi = lane >> 4;
  const _Float16* p = base + (size_t)(n0 + n) * ld + k0 + hi * 8;
  v16h b;
#pragma unroll
  for (int t = 0; t < 8; ++t) b[t] = p[t];
#pragma unroll
  for (int t = 0; t < 8; ++t) b[8 + t] = p[16 + t];
  return b;
}

// fp32 [K][N] -> f16 transposed [N][K]
__global__ __launch_bounds__(256) void cvt_t_kernel(const float* __restrict__ src,
                                                    _Float16* __restrict__ dst,
                                                    int K, int N) {
  size_t i = (size_t)blockIdx.x * 256 + threadIdx.x;
  if (i >= (size_t)K * N) return;
  size_t n = i / (size_t)K, k = i % (size_t)K;
  dst[n * (size_t)K + k] = (_Float16)src[k * (size_t)N + n];
}

// ---- per-entity transformer encoder ----
__global__ __launch_bounds__(256) void encode_kernel(
    const int* __restrict__ head_part, const int* __restrict__ tail_part,
    const int* __restrict__ nodes_idx, const int* __restrict__ hashes_idx,
    const float* __restrict__ node_emb, const float* __restrict__ node_proj_w,
    const float* __restrict__ node_proj_b, const float* __restrict__ anchor_emb,
    const float* __restrict__ type_emb,
    const _Float16* __restrict__ wqkv_t, const float* __restrict__ bqkv,
    const _Float16* __restrict__ wo_t,  const float* __restrict__ bo,
    const float* __restrict__ ln1_g, const float* __restrict__ ln1_b,
    const _Float16* __restrict__ w1_t, const float* __restrict__ b1,
    const _Float16* __restrict__ w2_t, const float* __restrict__ b2,
    const float* __restrict__ ln2_g, const float* __restrict__ ln2_b,
    float* __restrict__ enc) {
  __shared__ _Float16 sh_seq[32 * SEQ_LD];
  __shared__ _Float16 sh_o  [32 * SEQ_LD];
  __shared__ _Float16 sh_q  [32 * QK_LD];
  __shared__ _Float16 sh_k  [32 * QK_LD];
  __shared__ _Float16 sh_vt [128 * VT_LD];
  __shared__ float    sh_un [1024];          // scores (phase1) / LN stats (phase2+)
  _Float16* sh_p  = sh_q;                     // p reuses q storage (q dead after scores)
  float* sh_ps   = sh_un;                     // [32][8]
  float* sh_ps2  = sh_un + 256;               // [32][8]
  float* sh_mean = sh_un + 512;               // [32]
  float* sh_rstd = sh_un + 544;               // [32]
  float* sh_s    = sh_un;                     // [32][32] attention scores

  const int tid  = threadIdx.x;
  const int lane = tid & 31;
  // wave index is uniform: pin to SGPR so tile indices are scalar control flow.
  const int wv   = __builtin_amdgcn_readfirstlane(tid >> 5);
  const int m    = blockIdx.x;
  const int ent  = (m < 32) ? head_part[m * 3] : tail_part[m - 32];
  const int c0   = tid * 4;
  const v8f zf = {};

  // ---------- phase 0: build seq [32][1024] f16 ----------
  {
    int nidx[6];
#pragma unroll
    for (int t = 0; t < 6; ++t) nidx[t] = nodes_idx[(size_t)ent * 6 + t];
    float acc[6][4];
#pragma unroll
    for (int t = 0; t < 6; ++t)
#pragma unroll
      for (int c = 0; c < 4; ++c) acc[t][c] = 0.f;
    for (int i = 0; i < 32; ++i) {
      float w0 = node_proj_w[i * EDIM + c0 + 0];
      float w1v = node_proj_w[i * EDIM + c0 + 1];
      float w2v = node_proj_w[i * EDIM + c0 + 2];
      float w3 = node_proj_w[i * EDIM + c0 + 3];
#pragma unroll
      for (int t = 0; t < 6; ++t) {
        float nv = node_emb[(size_t)nidx[t] * 32 + i];
        acc[t][0] += nv * w0; acc[t][1] += nv * w1v;
        acc[t][2] += nv * w2v; acc[t][3] += nv * w3;
      }
    }
#pragma unroll
    for (int t = 0; t < 6; ++t) {
      int te = (t == 0) ? 0 : 1;
#pragma unroll
      for (int c = 0; c < 4; ++c) {
        float v = acc[t][c] + node_proj_b[c0 + c] + type_emb[te * EDIM + c0 + c];
        sh_seq[t * SEQ_LD + c0 + c] = (_Float16)v;
      }
    }
    for (int r = 0; r < 20; ++r) {
      int aidx = hashes_idx[(size_t)ent * 20 + r];
      const float* ap = anchor_emb + (size_t)aidx * EDIM;
#pragma unroll
      for (int c = 0; c < 4; ++c)
        sh_seq[(6 + r) * SEQ_LD + c0 + c] =
            (_Float16)(ap[c0 + c] + type_emb[2 * EDIM + c0 + c]);
    }
    for (int r = 26; r < 32; ++r)
#pragma unroll
      for (int c = 0; c < 4; ++c) sh_seq[r * SEQ_LD + c0 + c] = (_Float16)0.f;
  }
  __syncthreads();

  const int n16 = lane & 15, hi16 = lane >> 4;

  // ---------- phase 1: attention, head by head ----------
  const float inv_sqrt_dh = 0.0883883476483184f;
  for (int h = 0; h < 8; ++h) {
    // QKV: wave -> one row-tile (rt=wv&1), 6 (mat,ct) pairs; A broadcast across 6 WMMAs
    {
      const int rt = wv & 1;
      const int cs = (wv >> 1) * 6;       // 6 of 24 (mat,ct) pairs
      v8f qa[6];
#pragma unroll
      for (int j = 0; j < 6; ++j) qa[j] = zf;
      for (int ks = 0; ks < 32; ++ks) {
        v16h a = ldA(sh_seq, rt * 16, ks * 32, SEQ_LD);
#pragma unroll
        for (int j = 0; j < 6; ++j) {
          const int idx = cs + j, mat = idx >> 3, ct = idx & 7;
          const int ng = mat * 1024 + h * 128 + ct * 16;
          qa[j] = wmma_f16(a, ldB(wqkv_t, ng, ks * 32, 1024), qa[j]);
        }
      }
#pragma unroll
      for (int j = 0; j < 6; ++j) {
        const int idx = cs + j, mat = idx >> 3, ct = idx & 7;
        const int ng = mat * 1024 + h * 128 + ct * 16;
        float bias = bqkv[ng + n16];
        int colh = ct * 16 + n16;
        if (mat == 0) {
#pragma unroll
          for (int i = 0; i < 8; ++i)
            sh_q[(rt * 16 + hi16 * 8 + i) * QK_LD + colh] = (_Float16)(qa[j][i] + bias);
        } else if (mat == 1) {
#pragma unroll
          for (int i = 0; i < 8; ++i)
            sh_k[(rt * 16 + hi16 * 8 + i) * QK_LD + colh] = (_Float16)(qa[j][i] + bias);
        } else {
#pragma unroll
          for (int i = 0; i < 8; ++i)
            sh_vt[colh * VT_LD + rt * 16 + hi16 * 8 + i] = (_Float16)(qa[j][i] + bias);
        }
      }
    }
    __syncthreads();
    // scores = q @ k^T / sqrt(dh): 2x2 tiles, K=128
    if (wv < 4) {
      const int rt = wv >> 1, ct = wv & 1;
      v8f c = zf;
#pragma unroll
      for (int ks = 0; ks < 4; ++ks) {
        v16h a = ldA(sh_q, rt * 16, ks * 32, QK_LD);
        v16h b = ldB(sh_k, ct * 16, ks * 32, QK_LD);  // k rows act as B^T
        c = wmma_f16(a, b, c);
      }
#pragma unroll
      for (int i = 0; i < 8; ++i)
        sh_s[(rt * 16 + hi16 * 8 + i) * 32 + ct * 16 + n16] = c[i] * inv_sqrt_dh;
    }
    __syncthreads();
    // masked softmax (rows/cols >= 26 masked out), write p as f16
    if (tid < 32) {
      int row = tid;
      if (row < 26) {
        float sv[26], mx = -1e30f;
#pragma unroll
        for (int k = 0; k < 26; ++k) { sv[k] = sh_s[row * 32 + k]; mx = fmaxf(mx, sv[k]); }
        float sum = 0.f;
#pragma unroll
        for (int k = 0; k < 26; ++k) { sv[k] = __expf(sv[k] - mx); sum += sv[k]; }
        float inv = 1.f / sum;
#pragma unroll
        for (int k = 0; k < 26; ++k) sh_p[row * P_LD + k] = (_Float16)(sv[k] * inv);
#pragma unroll
        for (int k = 26; k < 32; ++k) sh_p[row * P_LD + k] = (_Float16)0.f;
      } else {
#pragma unroll
        for (int k = 0; k < 32; ++k) sh_p[row * P_LD + k] = (_Float16)0.f;
      }
    }
    __syncthreads();
    // o_head = p @ v: 2x8 tiles, single K=32 WMMA each
    for (int tt = wv; tt < 16; tt += 8) {
      const int rt = tt >> 3, ct = tt & 7;
      v16h a = ldA(sh_p, rt * 16, 0, P_LD);
      v16h b = ldB(sh_vt, ct * 16, 0, VT_LD);
      v8f c = wmma_f16(a, b, zf);
#pragma unroll
      for (int i = 0; i < 8; ++i)
        sh_o[(rt * 16 + hi16 * 8 + i) * SEQ_LD + h * 128 + ct * 16 + n16] = (_Float16)c[i];
    }
    __syncthreads();
  }

  // per-wave GEMM mapping for the 32x1024 GEMMs: one row-tile, 16 col-tiles
  const int rt2 = wv >> 2;            // 0..1
  const int cb  = (wv & 3) * 16;      // first of 16 col-tiles

  // ---------- phase 2: x = seq + o @ wo + bo (in place into sh_seq) ----------
  {
    v8f a2[16];
#pragma unroll
    for (int j = 0; j < 16; ++j) a2[j] = zf;
    for (int ks = 0; ks < 32; ++ks) {
      v16h a = ldA(sh_o, rt2 * 16, ks * 32, SEQ_LD);
#pragma unroll
      for (int j = 0; j < 16; ++j)
        a2[j] = wmma_f16(a, ldB(wo_t, (cb + j) * 16, ks * 32, 1024), a2[j]);
    }
#pragma unroll
    for (int j = 0; j < 16; ++j) {
      int colg = (cb + j) * 16 + n16;
      float bias = bo[colg];
#pragma unroll
      for (int i = 0; i < 8; ++i) {
        int idx = (rt2 * 16 + hi16 * 8 + i) * SEQ_LD + colg;
        sh_seq[idx] = (_Float16)(a2[j][i] + bias + (float)sh_seq[idx]);
      }
    }
  }
  __syncthreads();

  // row stats for LN1
  {
    int row = tid >> 3, sub = tid & 7;
    float s = 0.f, s2 = 0.f;
    for (int c = sub * 128; c < sub * 128 + 128; ++c) {
      float v = (float)sh_seq[row * SEQ_LD + c]; s += v; s2 += v * v;
    }
    sh_ps[row * 8 + sub] = s; sh_ps2[row * 8 + sub] = s2;
    __syncthreads();
    if (tid < 32) {
      float a = 0.f, b = 0.f;
#pragma unroll
      for (int j = 0; j < 8; ++j) { a += sh_ps[tid * 8 + j]; b += sh_ps2[tid * 8 + j]; }
      float mu = a * (1.f / 1024.f);
      sh_mean[tid] = mu;
      sh_rstd[tid] = rsqrtf(b * (1.f / 1024.f) - mu * mu + 1e-5f);
    }
    __syncthreads();
  }
  // apply LN1 (zero pad rows)
  {
    float g[4], bb[4];
#pragma unroll
    for (int c = 0; c < 4; ++c) { g[c] = ln1_g[c0 + c]; bb[c] = ln1_b[c0 + c]; }
    for (int row = 0; row < 32; ++row) {
      float mu = sh_mean[row], rs = sh_rstd[row];
#pragma unroll
      for (int c = 0; c < 4; ++c) {
        int idx = row * SEQ_LD + c0 + c;
        float v = (row < 26) ? (((float)sh_seq[idx] - mu) * rs * g[c] + bb[c]) : 0.f;
        sh_seq[idx] = (_Float16)v;
      }
    }
  }
  __syncthreads();

  // ---------- phase 3: FFN, split-K over 4 chunks, persistent accumulators ----------
  v8f acc[16];
#pragma unroll
  for (int j = 0; j < 16; ++j) acc[j] = zf;
  for (int cc4 = 0; cc4 < 4; ++cc4) {
    // h1 = relu(x @ w1 + b1) for this 1024-wide chunk -> sh_o (2 groups of 8 cols)
#pragma unroll 1
    for (int g = 0; g < 2; ++g) {
      v8f t8[8];
#pragma unroll
      for (int j = 0; j < 8; ++j) t8[j] = zf;
      for (int ks = 0; ks < 32; ++ks) {
        v16h a = ldA(sh_seq, rt2 * 16, ks * 32, SEQ_LD);
#pragma unroll
        for (int j = 0; j < 8; ++j)
          t8[j] = wmma_f16(a, ldB(w1_t, cc4 * 1024 + (cb + g * 8 + j) * 16, ks * 32, 1024), t8[j]);
      }
#pragma unroll
      for (int j = 0; j < 8; ++j) {
        int col = (cb + g * 8 + j) * 16 + n16;
        float bias = b1[cc4 * 1024 + col];
#pragma unroll
        for (int i = 0; i < 8; ++i)
          sh_o[(rt2 * 16 + hi16 * 8 + i) * SEQ_LD + col] =
              (_Float16)fmaxf(t8[j][i] + bias, 0.f);
      }
    }
    __syncthreads();
    // y += h1 @ w2[chunk rows]; A fragment broadcast to all 16 accumulators
    for (int ks = 0; ks < 32; ++ks) {
      v16h a = ldA(sh_o, rt2 * 16, ks * 32, SEQ_LD);
#pragma unroll
      for (int j = 0; j < 16; ++j)
        acc[j] = wmma_f16(a, ldB(w2_t, (cb + j) * 16, cc4 * 1024 + ks * 32, 4096), acc[j]);
    }
    __syncthreads();
  }
  // epilogue: pre-LN2 = y + b2 + x  -> sh_o
#pragma unroll
  for (int j = 0; j < 16; ++j) {
    int colg = (cb + j) * 16 + n16;
    float bias = b2[colg];
#pragma unroll
    for (int i = 0; i < 8; ++i) {
      int row = rt2 * 16 + hi16 * 8 + i;
      float v = acc[j][i] + bias + (float)sh_seq[row * SEQ_LD + colg];
      sh_o[row * SEQ_LD + colg] = (row < 26) ? (_Float16)v : (_Float16)0.f;
    }
  }
  __syncthreads();

  // row stats for LN2
  {
    int row = tid >> 3, sub = tid & 7;
    float s = 0.f, s2 = 0.f;
    for (int c = sub * 128; c < sub * 128 + 128; ++c) {
      float v = (float)sh_o[row * SEQ_LD + c]; s += v; s2 += v * v;
    }
    sh_ps[row * 8 + sub] = s; sh_ps2[row * 8 + sub] = s2;
    __syncthreads();
    if (tid < 32) {
      float a = 0.f, b = 0.f;
#pragma unroll
      for (int j = 0; j < 8; ++j) { a += sh_ps[tid * 8 + j]; b += sh_ps2[tid * 8 + j]; }
      float mu = a * (1.f / 1024.f);
      sh_mean[tid] = mu;
      sh_rstd[tid] = rsqrtf(b * (1.f / 1024.f) - mu * mu + 1e-5f);
    }
    __syncthreads();
  }
  // fused LN2 + mean pool over 26 tokens
#pragma unroll
  for (int c = 0; c < 4; ++c) {
    int col = c0 + c;
    float s = 0.f;
    for (int t = 0; t < 26; ++t)
      s += ((float)sh_o[t * SEQ_LD + col] - sh_mean[t]) * sh_rstd[t];
    enc[(size_t)m * EDIM + col] = s * ln2_g[col] * (1.f / 26.f) + ln2_b[col];
  }
}

// ---- RotatE scoring: one wave per (b, neg) pair ----
__global__ __launch_bounds__(256) void score_kernel(const float* __restrict__ enc,
                                                    const float* __restrict__ rel_emb,
                                                    const int* __restrict__ head_part,
                                                    float* __restrict__ out) {
  int lane = threadIdx.x & 31;
  int pair = blockIdx.x * 8 + __builtin_amdgcn_readfirstlane(threadIdx.x >> 5);
  int b = pair >> 5;
  const float* hd = enc + (size_t)b * EDIM;
  const float* tl = enc + (size_t)(32 + pair) * EDIM;
  const float* rl = rel_emb + (size_t)head_part[b * 3 + 1] * 512;
  const float ps = 134.04128655773344f;  // pi / ((GAMMA+2)/HID)
  float s = 0.f;
  for (int d = lane; d < 512; d += 32) {
    float sr, cr;
    __sincosf(rl[d] * ps, &sr, &cr);
    float rh = hd[d], ih = hd[512 + d], rt = tl[d], it = tl[512 + d];
    float rs = rh * cr - ih * sr - rt;
    float is = rh * sr + ih * cr - it;
    s += sqrtf(rs * rs + is * is);
  }
#pragma unroll
  for (int off = 16; off; off >>= 1) s += __shfl_down(s, off);
  if (lane == 0) out[pair] = 10.0f - s;
}

extern "C" void kernel_launch(void* const* d_in, const int* in_sizes, int n_in,
                              void* d_out, int out_size, void* d_ws, size_t ws_size,
                              hipStream_t stream) {
  const int*   head_part   = (const int*)  d_in[0];
  const int*   tail_part   = (const int*)  d_in[1];
  const int*   nodes_idx   = (const int*)  d_in[2];
  const int*   hashes_idx  = (const int*)  d_in[3];
  const float* node_emb    = (const float*)d_in[4];
  const float* node_proj_w = (const float*)d_in[5];
  const float* node_proj_b = (const float*)d_in[6];
  const float* anchor_emb  = (const float*)d_in[7];
  const float* type_emb    = (const float*)d_in[8];
  const float* rel_emb     = (const float*)d_in[9];
  const float* wqkv        = (const float*)d_in[10];
  const float* bqkv        = (const float*)d_in[11];
  const float* wo          = (const float*)d_in[12];
  const float* bo          = (const float*)d_in[13];
  const float* ln1_g       = (const float*)d_in[14];
  const float* ln1_b       = (const float*)d_in[15];
  const float* w1          = (const float*)d_in[16];
  const float* b1          = (const float*)d_in[17];
  const float* w2          = (const float*)d_in[18];
  const float* b2          = (const float*)d_in[19];
  const float* ln2_g       = (const float*)d_in[20];
  const float* ln2_b       = (const float*)d_in[21];

  char* ws = (char*)d_ws;
  size_t off = 0;
  _Float16* wqkv_t = (_Float16*)(ws + off); off += (size_t)3072 * 1024 * 2;
  _Float16* wo_t   = (_Float16*)(ws + off); off += (size_t)1024 * 1024 * 2;
  _Float16* w1_t   = (_Float16*)(ws + off); off += (size_t)4096 * 1024 * 2;
  _Float16* w2_t   = (_Float16*)(ws + off); off += (size_t)1024 * 4096 * 2;
  float*    enc    = (float*)   (ws + off); off += (size_t)1056 * 1024 * 4;

  cvt_t_kernel<<<(3072 * 1024 + 255) / 256, 256, 0, stream>>>(wqkv, wqkv_t, 1024, 3072);
  cvt_t_kernel<<<(1024 * 1024 + 255) / 256, 256, 0, stream>>>(wo,   wo_t,   1024, 1024);
  cvt_t_kernel<<<(4096 * 1024 + 255) / 256, 256, 0, stream>>>(w1,   w1_t,   1024, 4096);
  cvt_t_kernel<<<(4096 * 1024 + 255) / 256, 256, 0, stream>>>(w2,   w2_t,   4096, 1024);

  encode_kernel<<<1056, 256, 0, stream>>>(
      head_part, tail_part, nodes_idx, hashes_idx, node_emb, node_proj_w,
      node_proj_b, anchor_emb, type_emb, wqkv_t, bqkv, wo_t, bo, ln1_g, ln1_b,
      w1_t, b1, w2_t, b2, ln2_g, ln2_b, enc);

  score_kernel<<<128, 256, 0, stream>>>(enc, rel_emb, head_part, (float*)d_out);
}
